// RecurrentGNN_79491254714579
// MI455X (gfx1250) — matrix operand order repaired and checked by reading, not compiled, not run on hardware
//
#include <hip/hip_runtime.h>
#include <hip/hip_bf16.h>
#include <math.h>

// ---- problem constants (match reference) ----
#define NB   32          // batch
#define NN   100         // nodes
#define NH   128         // hidden = input size
#define NE   9900        // edges (N*(N-1))
#define NREL 135         // REL + D = 7 + 128
#define NODES (NB*NN)    // 3200 rows for node-level GEMMs
#define SBUF  (NODES*NH) // 409600 floats per node buffer

typedef __attribute__((ext_vector_type(16))) __bf16 v16bf;
typedef __attribute__((ext_vector_type(8)))  float  v8f;
typedef __attribute__((ext_vector_type(4)))  unsigned int u32x4;

union BF16x16 { unsigned short u[16]; u32x4 q[2]; v16bf v; };

__device__ __forceinline__ unsigned short f2bf(float f) {
    unsigned u = __builtin_bit_cast(unsigned, f);
    unsigned r = u + 0x7FFFu + ((u >> 16) & 1u);   // round-to-nearest-even
    return (unsigned short)(r >> 16);
}

__device__ __forceinline__ v8f wmma_bf16(v16bf a, v16bf b, v8f c) {
    return __builtin_amdgcn_wmma_f32_16x16x32_bf16(false, a, false, b, (short)0, c, false, false);
}

// A fragment (16x32) from LDS bf16 tile with row stride `stride` (elements)
__device__ __forceinline__ v16bf load_a_lds(const unsigned short* As, int stride, int k0, int lane) {
    int row = lane & 15;
    int kb  = (lane >> 4) * 16;
    const unsigned short* p = As + row * stride + k0 + kb;
    BF16x16 r;
#pragma unroll
    for (int j = 0; j < 8; ++j) { r.u[j] = p[j]; r.u[j + 8] = p[8 + j]; }
    return r.v;
}

// B fragment (32x16) from packed bf16 weight: layout [ktile][col(128)][32 k]
__device__ __forceinline__ v16bf load_b_packed(const unsigned short* __restrict__ Wb,
                                               int kt, int gcol0, int lane) {
    const unsigned short* p = Wb + ((size_t)kt << 12) + ((gcol0 + (lane & 15)) << 5) + ((lane >> 4) << 4);
    BF16x16 r;
    r.q[0] = ((const u32x4*)p)[0];
    r.q[1] = ((const u32x4*)p)[1];
    return r.v;
}

// cooperative stage of a contiguous 16x128 f32 tile into LDS bf16
__device__ __forceinline__ void stage_tile(const float* __restrict__ src,
                                           unsigned short* dst, int tid) {
    for (int idx = tid; idx < 16 * NH; idx += 256) dst[idx] = f2bf(src[idx]);
}

// ---------------------------------------------------------------------------
// Pack all weight matrices (f32 [K x 128]) into bf16 fragment layout
// dst[kt*4096 + c*32 + kk] = bf16(W[(kt*32+kk)*128 + c]), zero-padded past K
// ---------------------------------------------------------------------------
#define NW 15
struct PackArgs {
    const float*    src[NW];
    unsigned short* dst[NW];
    int             K[NW];
};

__global__ void pack_weights_k(PackArgs pa) {
    const int w = blockIdx.y;
    const int K = pa.K[w];
    const int total = ((K + 31) >> 5) << 12;   // ktiles * 4096
    const float* __restrict__ src = pa.src[w];
    unsigned short* __restrict__ dst = pa.dst[w];
    for (int idx = blockIdx.x * blockDim.x + threadIdx.x; idx < total;
         idx += gridDim.x * blockDim.x) {
        int kt  = idx >> 12;
        int rem = idx & 4095;
        int c   = rem >> 5;
        int kk  = rem & 31;
        int k   = (kt << 5) + kk;
        dst[idx] = (k < K) ? f2bf(src[(size_t)k * NH + c]) : (unsigned short)0;
    }
}

// ---------------------------------------------------------------------------
// node_pre: per 16-row tile (8 waves):
//   Hr = hidden @ msg_w1_top,  Hs = hidden @ msg_w1_bot     (f32 out)
//   res = relu(relu(inputs @ res_w1 + b1) @ res_w2 + b2)    (f32 out)
// ---------------------------------------------------------------------------
__global__ void node_pre_k(const float* __restrict__ hidden, const float* __restrict__ inputs,
                           const unsigned short* __restrict__ msgw1p,
                           const unsigned short* __restrict__ resw1p, const float* __restrict__ res_b1,
                           const unsigned short* __restrict__ resw2p, const float* __restrict__ res_b2,
                           float* __restrict__ Hr, float* __restrict__ Hs, float* __restrict__ res)
{
    __shared__ unsigned short hidS[16 * NH];
    __shared__ unsigned short inS[16 * NH];
    __shared__ unsigned short r1S[16 * NH];
    const int tid = threadIdx.x, wave = tid >> 5, lane = tid & 31, gcol0 = wave * 16;
    const long row0 = (long)blockIdx.x * 16;
    stage_tile(hidden + row0 * NH, hidS, tid);
    stage_tile(inputs + row0 * NH, inS, tid);
    __syncthreads();

    v8f cHr = {}, cHs = {}, cR = {};
#pragma unroll
    for (int kt = 0; kt < 4; ++kt) {
        v16bf ah = load_a_lds(hidS, NH, kt * 32, lane);
        cHr = wmma_bf16(ah, load_b_packed(msgw1p, kt,     gcol0, lane), cHr);
        cHs = wmma_bf16(ah, load_b_packed(msgw1p, kt + 4, gcol0, lane), cHs);
        v16bf ai = load_a_lds(inS, NH, kt * 32, lane);
        cR  = wmma_bf16(ai, load_b_packed(resw1p, kt, gcol0, lane), cR);
    }
    const int col = gcol0 + (lane & 15);
    const int rb  = (lane >> 4) * 8;
#pragma unroll
    for (int v = 0; v < 8; ++v) {
        long m = row0 + rb + v;
        Hr[m * NH + col] = cHr[v];
        Hs[m * NH + col] = cHs[v];
        r1S[(rb + v) * NH + col] = f2bf(fmaxf(cR[v] + res_b1[col], 0.f));
    }
    __syncthreads();
    v8f cR2 = {};
#pragma unroll
    for (int kt = 0; kt < 4; ++kt)
        cR2 = wmma_bf16(load_a_lds(r1S, NH, kt * 32, lane),
                        load_b_packed(resw2p, kt, gcol0, lane), cR2);
#pragma unroll
    for (int v = 0; v < 8; ++v) {
        long m = row0 + rb + v;
        res[m * NH + col] = fmaxf(cR2[v] + res_b2[col], 0.f);
    }
}

// ---------------------------------------------------------------------------
// edge_msg: one block per (b, recv); factorized concat-GEMM layer 1
//   hidden_node[b,recv] = mean_s tanh(tanh(Hr[b,recv]+Hs[b,s]+b1) @ W2 + b2)
// ---------------------------------------------------------------------------
__global__ void edge_msg_kernel(const float* __restrict__ Hr, const float* __restrict__ Hs,
                                const float* __restrict__ b1,
                                const unsigned short* __restrict__ W2p, const float* __restrict__ b2,
                                float* __restrict__ hidden_node)
{
    __shared__ unsigned short As[16 * NH];
    const int br = blockIdx.x;
    const int b  = br / NN, r = br % NN;
    const int tid = threadIdx.x, wave = tid >> 5, lane = tid & 31, gcol0 = wave * 16;
    const float* HrRow = Hr + (long)br * NH;

    float colacc = 0.f;
    for (int t = 0; t < 7; ++t) {
        __syncthreads();
        for (int idx = tid; idx < 16 * NH; idx += 256) {
            int j  = idx >> 7, cc = idx & 127;
            int jr = t * 16 + j;
            int s  = (jr < 99) ? (jr + (jr >= r ? 1 : 0)) : 0;
            float v = HrRow[cc] + Hs[((long)b * NN + s) * NH + cc] + b1[cc];
            As[idx] = f2bf(tanhf(v));
        }
        __syncthreads();
        v8f c = {};
#pragma unroll
        for (int kt = 0; kt < 4; ++kt)
            c = wmma_bf16(load_a_lds(As, NH, kt * 32, lane),
                          load_b_packed(W2p, kt, gcol0, lane), c);
        const int col = gcol0 + (lane & 15);
        const int rb  = (lane >> 4) * 8;
#pragma unroll
        for (int v = 0; v < 8; ++v) {
            int jr = t * 16 + rb + v;
            if (jr < 99) colacc += tanhf(c[v] + b2[col]);
        }
    }
    float other = __shfl_xor(colacc, 16, 32);
    if (lane < 16)
        hidden_node[(long)br * NH + gcol0 + lane] = (colacc + other) * (1.0f / 99.0f);
}

// ---------------------------------------------------------------------------
// edge_pm: one block per (b, recv); edge_attr staged once to LDS, K padded 135->160
//   pm_node[b,recv] = mean_e relu(relu(ea @ W1 + b1) @ W2 + b2)
// ---------------------------------------------------------------------------
__global__ void edge_pm_kernel(const float* __restrict__ EA,
                               const unsigned short* __restrict__ W1p, const float* __restrict__ b1,
                               const unsigned short* __restrict__ W2p, const float* __restrict__ b2,
                               float* __restrict__ pm_node)
{
    __shared__ unsigned short Ea_s[16 * 160];
    __shared__ unsigned short As[16 * NH];
    const int br = blockIdx.x;
    const int b  = br / NN, r = br % NN;
    const int tid = threadIdx.x, wave = tid >> 5, lane = tid & 31, gcol0 = wave * 16;

    float colacc = 0.f;
    for (int t = 0; t < 7; ++t) {
        __syncthreads();
        for (int idx = tid; idx < 16 * 160; idx += 256) {
            int j = idx / 160, k = idx % 160;
            int jr = t * 16 + j;
            unsigned short v = 0;
            if (jr < 99 && k < NREL) {
                int s   = jr + (jr >= r ? 1 : 0);
                int pos = (r < s) ? r : r - 1;
                long e  = (long)s * 99 + pos;
                v = f2bf(EA[((long)b * NE + e) * NREL + k]);
            }
            Ea_s[idx] = v;
        }
        // prefetch next tile's gathered edge rows (strided gather; HBM stream)
        if (t < 6) {
            for (int idx = tid; idx < 16 * 5; idx += 256) {
                int j = idx / 5, chunk = idx % 5;
                int jr = (t + 1) * 16 + j;
                if (jr < 99) {
                    int s   = jr + (jr >= r ? 1 : 0);
                    int pos = (r < s) ? r : r - 1;
                    long e  = (long)s * 99 + pos;
                    __builtin_prefetch(EA + ((long)b * NE + e) * NREL + chunk * 28, 0, 1);
                }
            }
        }
        __syncthreads();
        // layer 1: 16x160(pad) @ 160x128, zero-padded on both sides
        v8f c1 = {};
#pragma unroll
        for (int kt = 0; kt < 5; ++kt)
            c1 = wmma_bf16(load_a_lds(Ea_s, 160, kt * 32, lane),
                           load_b_packed(W1p, kt, gcol0, lane), c1);
        const int col = gcol0 + (lane & 15);
        const int rb  = (lane >> 4) * 8;
#pragma unroll
        for (int v = 0; v < 8; ++v)
            As[(rb + v) * NH + col] = f2bf(fmaxf(c1[v] + b1[col], 0.f));
        __syncthreads();
        // layer 2: 16x128 @ 128x128
        v8f c2 = {};
#pragma unroll
        for (int kt = 0; kt < 4; ++kt)
            c2 = wmma_bf16(load_a_lds(As, NH, kt * 32, lane),
                           load_b_packed(W2p, kt, gcol0, lane), c2);
#pragma unroll
        for (int v = 0; v < 8; ++v) {
            int jr = t * 16 + rb + v;
            if (jr < 99) colacc += fmaxf(c2[v] + b2[col], 0.f);
        }
    }
    float other = __shfl_xor(colacc, 16, 32);
    if (lane < 16)
        pm_node[(long)br * NH + gcol0 + lane] = (colacc + other) * (1.0f / 99.0f);
}

// ---------------------------------------------------------------------------
// node_post: per 16-row tile; fused GRU gates + combine + 3-layer out-MLP.
// Gate accumulators share the C-fragment layout, so r/i/n combine lane-locally.
// ---------------------------------------------------------------------------
__global__ void node_post_k(const float* __restrict__ pm_node, const float* __restrict__ res,
                            const float* __restrict__ hid_node, const float* __restrict__ hidden,
                            const unsigned short* __restrict__ irp, const float* __restrict__ ir_b,
                            const unsigned short* __restrict__ hrp,
                            const unsigned short* __restrict__ iip, const float* __restrict__ ii_b,
                            const unsigned short* __restrict__ hip_,
                            const unsigned short* __restrict__ inp, const float* __restrict__ in_b,
                            const unsigned short* __restrict__ hhp,
                            const unsigned short* __restrict__ ow1p, const float* __restrict__ ob1,
                            const unsigned short* __restrict__ ow2p, const float* __restrict__ ob2,
                            const unsigned short* __restrict__ ow3p, const float* __restrict__ ob3,
                            float* __restrict__ hnew_out, float* __restrict__ pred)
{
    __shared__ unsigned short preS[16 * NH];   // present -> hnew -> o2
    __shared__ unsigned short hidS[16 * NH];   // hidden_node -> o1
    const int tid = threadIdx.x, wave = tid >> 5, lane = tid & 31, gcol0 = wave * 16;
    const long row0 = (long)blockIdx.x * 16;
    const float* pmT = pm_node + row0 * NH;
    const float* reT = res + row0 * NH;
    const float* hnT = hid_node + row0 * NH;
    for (int idx = tid; idx < 16 * NH; idx += 256) {
        preS[idx] = f2bf(pmT[idx] + reT[idx]);   // present = pm_node + res
        hidS[idx] = f2bf(hnT[idx]);
    }
    __syncthreads();

    v8f cr = {}, ci = {}, cn1 = {}, cn2 = {};
#pragma unroll
    for (int kt = 0; kt < 4; ++kt) {
        v16bf ap = load_a_lds(preS, NH, kt * 32, lane);
        v16bf ah = load_a_lds(hidS, NH, kt * 32, lane);
        cr  = wmma_bf16(ap, load_b_packed(irp,  kt, gcol0, lane), cr);
        cr  = wmma_bf16(ah, load_b_packed(hrp,  kt, gcol0, lane), cr);
        ci  = wmma_bf16(ap, load_b_packed(iip,  kt, gcol0, lane), ci);
        ci  = wmma_bf16(ah, load_b_packed(hip_, kt, gcol0, lane), ci);
        cn1 = wmma_bf16(ap, load_b_packed(inp,  kt, gcol0, lane), cn1);
        cn2 = wmma_bf16(ah, load_b_packed(hhp,  kt, gcol0, lane), cn2);
    }
    const int col = gcol0 + (lane & 15);
    const int rb  = (lane >> 4) * 8;
    __syncthreads();                           // everyone done reading preS/hidS
#pragma unroll
    for (int v = 0; v < 8; ++v) {
        long m = row0 + rb + v;
        float rg = 1.f / (1.f + __expf(-(cr[v] + ir_b[col])));
        float ig = 1.f / (1.f + __expf(-(ci[v] + ii_b[col])));
        float ng = tanhf(cn1[v] + in_b[col] + rg * cn2[v]);
        float h  = (1.f - ig) * ng + ig * hidden[m * NH + col];
        hnew_out[m * NH + col] = h;
        preS[(rb + v) * NH + col] = f2bf(h);   // hnew tile
    }
    __syncthreads();
    v8f c1 = {};
#pragma unroll
    for (int kt = 0; kt < 4; ++kt)
        c1 = wmma_bf16(load_a_lds(preS, NH, kt * 32, lane),
                       load_b_packed(ow1p, kt, gcol0, lane), c1);
#pragma unroll
    for (int v = 0; v < 8; ++v)
        hidS[(rb + v) * NH + col] = f2bf(fmaxf(c1[v] + ob1[col], 0.f));  // o1 tile
    __syncthreads();
    v8f c2 = {};
#pragma unroll
    for (int kt = 0; kt < 4; ++kt)
        c2 = wmma_bf16(load_a_lds(hidS, NH, kt * 32, lane),
                       load_b_packed(ow2p, kt, gcol0, lane), c2);
    __syncthreads();                           // preS (hnew) fully consumed
#pragma unroll
    for (int v = 0; v < 8; ++v)
        preS[(rb + v) * NH + col] = f2bf(fmaxf(c2[v] + ob2[col], 0.f));  // o2 tile
    __syncthreads();
    v8f c3 = {};
#pragma unroll
    for (int kt = 0; kt < 4; ++kt)
        c3 = wmma_bf16(load_a_lds(preS, NH, kt * 32, lane),
                       load_b_packed(ow3p, kt, gcol0, lane), c3);
#pragma unroll
    for (int v = 0; v < 8; ++v) {
        long m = row0 + rb + v;
        pred[m * NH + col] = c3[v] + ob3[col];
    }
}

// ---------------------------------------------------------------------------
extern "C" void kernel_launch(void* const* d_in, const int* in_sizes, int n_in,
                              void* d_out, int out_size, void* d_ws, size_t ws_size,
                              hipStream_t stream)
{
    const float* inputs    = (const float*)d_in[0];
    const float* edge_attr = (const float*)d_in[1];
    const float* hidden    = (const float*)d_in[2];
    const float* msg_w1 = (const float*)d_in[3];  const float* msg_b1 = (const float*)d_in[4];
    const float* msg_w2 = (const float*)d_in[5];  const float* msg_b2 = (const float*)d_in[6];
    const float* pm_w1  = (const float*)d_in[7];  const float* pm_b1  = (const float*)d_in[8];
    const float* pm_w2  = (const float*)d_in[9];  const float* pm_b2  = (const float*)d_in[10];
    const float* res_w1 = (const float*)d_in[11]; const float* res_b1 = (const float*)d_in[12];
    const float* res_w2 = (const float*)d_in[13]; const float* res_b2 = (const float*)d_in[14];
    const float* ir_w = (const float*)d_in[15];   const float* ir_b = (const float*)d_in[16];
    const float* ii_w = (const float*)d_in[17];   const float* ii_b = (const float*)d_in[18];
    const float* in_w = (const float*)d_in[19];   const float* in_b = (const float*)d_in[20];
    const float* hr_w = (const float*)d_in[21];
    const float* hi_w = (const float*)d_in[22];
    const float* hh_w = (const float*)d_in[23];
    const float* out_w1 = (const float*)d_in[24]; const float* out_b1 = (const float*)d_in[25];
    const float* out_w2 = (const float*)d_in[26]; const float* out_b2 = (const float*)d_in[27];
    const float* out_w3 = (const float*)d_in[28]; const float* out_b3 = (const float*)d_in[29];

    float* ws = (float*)d_ws;
    float* Hr       = ws + (size_t)SBUF * 0;
    float* Hs       = ws + (size_t)SBUF * 1;
    float* hid_node = ws + (size_t)SBUF * 2;
    float* pm_node  = ws + (size_t)SBUF * 3;
    float* res      = ws + (size_t)SBUF * 4;
    unsigned short* pk = (unsigned short*)(ws + (size_t)SBUF * 5);

    // packed weight layout (ushort offsets, each a multiple of 4096)
    unsigned short* msgw1p = pk;                 // K=256 -> 8 tiles
    unsigned short* msgw2p = pk + 32768;         // 4 tiles
    unsigned short* pmw1p  = pk + 49152;         // K=135 -> 5 tiles (padded)
    unsigned short* pmw2p  = pk + 69632;
    unsigned short* resw1p = pk + 86016;
    unsigned short* resw2p = pk + 102400;
    unsigned short* irp    = pk + 118784;
    unsigned short* hrp    = pk + 135168;
    unsigned short* iip    = pk + 151552;
    unsigned short* hip_   = pk + 167936;
    unsigned short* inp    = pk + 184320;
    unsigned short* hhp    = pk + 200704;
    unsigned short* ow1p   = pk + 217088;
    unsigned short* ow2p   = pk + 233472;
    unsigned short* ow3p   = pk + 249856;

    float* pred = (float*)d_out;     // [B,N,D]
    float* hnew = pred + SBUF;       // [B,N,H]

    PackArgs pa;
    const float* srcs[NW] = { msg_w1, msg_w2, pm_w1, pm_w2, res_w1, res_w2,
                              ir_w, hr_w, ii_w, hi_w, in_w, hh_w,
                              out_w1, out_w2, out_w3 };
    unsigned short* dsts[NW] = { msgw1p, msgw2p, pmw1p, pmw2p, resw1p, resw2p,
                                 irp, hrp, iip, hip_, inp, hhp,
                                 ow1p, ow2p, ow3p };
    int Ks[NW] = { 256, 128, 135, 128, 128, 128, 128, 128, 128, 128, 128, 128,
                   128, 128, 128 };
    for (int i = 0; i < NW; ++i) { pa.src[i] = srcs[i]; pa.dst[i] = dsts[i]; pa.K[i] = Ks[i]; }

    pack_weights_k<<<dim3(32, NW), 256, 0, stream>>>(pa);

    node_pre_k<<<NODES / 16, 256, 0, stream>>>(hidden, inputs, msgw1p,
                                               resw1p, res_b1, resw2p, res_b2,
                                               Hr, Hs, res);

    edge_msg_kernel<<<NODES, 256, 0, stream>>>(Hr, Hs, msg_b1, msgw2p, msg_b2, hid_node);
    edge_pm_kernel<<<NODES, 256, 0, stream>>>(edge_attr, pmw1p, pm_b1, pmw2p, pm_b2, pm_node);

    node_post_k<<<NODES / 16, 256, 0, stream>>>(pm_node, res, hid_node, hidden,
                                                irp, ir_b, hrp,
                                                iip, ii_b, hip_,
                                                inp, in_b, hhp,
                                                ow1p, out_b1, ow2p, out_b2, ow3p, out_b3,
                                                hnew, pred);
}
